// CaiQuantLinear_53781580480832
// MI455X (gfx1250) — matrix operand order repaired and checked by reading, not compiled, not run on hardware
//
#include <hip/hip_runtime.h>
#include <cstdint>

#define INF       8192
#define OUTF      8192
#define TOKENS    32
#define GROUPSIZE 128

typedef __attribute__((ext_vector_type(16))) _Float16 v16h;
typedef __attribute__((ext_vector_type(8)))  _Float16 v8h;
typedef __attribute__((ext_vector_type(2)))  _Float16 h2;
typedef __attribute__((ext_vector_type(8)))  float    v8f;

// Dequant two 4-bit values (nibbles at sh, sh+4) into packed fp16:
//   bits | 0x6400 == fp16(1024 + q)  ->  fma: (1024+q)*s + (-(1024+z)*s) == (q-z)*s
static __device__ inline h2 dq_pair(uint32_t q, int sh, h2 s2, h2 c2) {
  uint32_t bits = ((q >> sh) & 0xFu) | (((q >> (sh + 4)) & 0xFu) << 16) | 0x64006400u;
  h2 v = __builtin_bit_cast(h2, bits);
  return v * s2 + c2;
}

// gfx1250 async global->LDS copy (ASYNCcnt-tracked, no VGPR round trip).
// VDST carries the LDS byte offset; low 32 bits of a generic LDS pointer are
// exactly that offset (ISA flat-LDS aperture).
static __device__ inline void async_copy_b128(uint32_t lds_byte_off, const void* gaddr) {
  asm volatile("global_load_async_to_lds_b128 %0, %1, off"
               :: "v"(lds_byte_off), "v"(gaddr) : "memory");
}
static __device__ inline void wait_asynccnt0() {
  asm volatile("s_wait_asynccnt 0x0" ::: "memory");
}

// One wave: 16 output columns x 32 rows (two 16x16x32 WMMA accumulators
// sharing one dequantized B fragment). Block: 8 waves -> 128 columns.
// grid.y = split-K chunk of KCHUNK. x slab staged in LDS once per block.
template <int KCHUNK>
__global__ __launch_bounds__(256) void gptq_wmma_gemm(
    const _Float16* __restrict__ x,        // [TOKENS, INF] fp16
    const uint32_t* __restrict__ qweight,  // [INF/8, OUTF] 8x4b packed along K
    const uint32_t* __restrict__ qzeros,   // [INF/GROUPSIZE, OUTF/8]
    const _Float16* __restrict__ scales,   // [INF/GROUPSIZE, OUTF]
    float* __restrict__ partial)           // [nsplit][TOKENS][OUTF] f32
{
  // +8 halfs (16B) row pad: rows shift 4 banks each -> every bank hit exactly
  // twice per 32-lane ds_load_b128 (optimal), 16B alignment preserved.
  constexpr int PITCH = KCHUNK + 8;
  __shared__ __align__(16) _Float16 sA[TOKENS * PITCH];

  const int lane = threadIdx.x & 31;
  const int wave = threadIdx.x >> 5;
  const int n    = blockIdx.x * 128 + wave * 16 + (lane & 15);
  const int kh   = lane >> 4;              // K-half select per WMMA lane layout
  const int kBeg = blockIdx.y * KCHUNK;

  // ---- Stage x slab [32 x KCHUNK] into LDS via async b128 copies ----
  constexpr int CPR    = KCHUNK / 8;       // 16B chunks per row
  constexpr int NCHUNK = TOKENS * CPR;     // <= 32 per thread => ASYNCcnt <= 32
  for (int c = threadIdx.x; c < NCHUNK; c += 256) {
    const int row = c / CPR;
    const int col = c - row * CPR;
    const _Float16* g = x + (size_t)row * INF + kBeg + col * 8;
    const uint32_t  l = (uint32_t)(uintptr_t)&sA[row * PITCH + col * 8];
    async_copy_b128(l, g);
  }
  wait_asynccnt0();
  __syncthreads();

  v8f acc0 = {};                           // rows 0..15
  v8f acc1 = {};                           // rows 16..31

  const int g0 = kBeg / GROUPSIZE;
  const int g1 = g0 + KCHUNK / GROUPSIZE;

  for (int g = g0; g < g1; ++g) {
    // Per-group dequant constants for this lane's column n.
    const float    s  = (float)scales[(size_t)g * OUTF + n];
    const uint32_t qz = qzeros[(size_t)g * (OUTF / 8) + (n >> 3)];
    const int      zp = (int)((qz >> ((n & 7) * 4)) & 0xFu) + 1;
    const _Float16 sh_ = (_Float16)s;
    const _Float16 ch_ = (_Float16)(-(float)(1024 + zp) * s);
    const h2 s2 = { sh_, sh_ };
    const h2 c2 = { ch_, ch_ };

#pragma unroll
    for (int kk = 0; kk < GROUPSIZE; kk += 32) {
      const int k0   = g * GROUPSIZE + kk;
      const int krel = k0 - kBeg;

      // ---- A fragments from LDS (16-bit A 16x32 WMMA layout) ----
      // lane<16: M=lane, K={0..7,16..23}; lane>=16: M=lane-16, K={8..15,24..31}
      const _Float16* a0p = &sA[(lane & 15) * PITCH + krel + kh * 8];
      const _Float16* a1p = a0p + 16 * PITCH;
      v8h a0lo = *(const v8h*)a0p;
      v8h a0hi = *(const v8h*)(a0p + 16);
      v8h a1lo = *(const v8h*)a1p;
      v8h a1hi = *(const v8h*)(a1p + 16);
      v16h a0, a1;
#pragma unroll
      for (int i = 0; i < 8; ++i) {
        a0[i] = a0lo[i]; a0[8 + i] = a0hi[i];
        a1[i] = a1lo[i]; a1[8 + i] = a1hi[i];
      }

      // ---- B tile 32x16: lane's 16 K values live in 2 packed dwords ----
      // Stream qweight non-temporally: read-once, keep x/scales L2-resident.
      const uint32_t* qp = qweight + (size_t)(k0 / 8 + kh) * OUTF + n;
      const uint32_t q0 = __builtin_nontemporal_load(qp);
      const uint32_t q1 = __builtin_nontemporal_load(qp + 2 * OUTF);

      v16h b;
#pragma unroll
      for (int i = 0; i < 4; ++i) {
        h2 p0 = dq_pair(q0, 8 * i, s2, c2);
        h2 p1 = dq_pair(q1, 8 * i, s2, c2);
        b[2 * i]     = p0[0]; b[2 * i + 1]     = p0[1];
        b[8 + 2 * i] = p1[0]; b[8 + 2 * i + 1] = p1[1];
      }

      acc0 = __builtin_amdgcn_wmma_f32_16x16x32_f16(false, a0, false, b,
                                                    (short)0, acc0, false, false);
      acc1 = __builtin_amdgcn_wmma_f32_16x16x32_f16(false, a1, false, b,
                                                    (short)0, acc1, false, false);
    }
  }

  // D layout: VGPR r -> M = r + 8*kh (acc0), +16 for acc1; N = lane&15.
  float* wp = partial + (size_t)blockIdx.y * TOKENS * OUTF + n;
  const int r0 = kh * 8;
#pragma unroll
  for (int r = 0; r < 8; ++r) {
    wp[(size_t)(r0 + r) * OUTF]      = acc0[r];
    wp[(size_t)(r0 + r + 16) * OUTF] = acc1[r];
  }
}

// Deterministic split-K reduction + bias + fp16 convert.
__global__ __launch_bounds__(256) void gptq_reduce_bias(
    const float* __restrict__ partial, const _Float16* __restrict__ bias,
    _Float16* __restrict__ out, int nsplit)
{
  const int idx = blockIdx.x * 256 + threadIdx.x;   // idx = m*OUTF + n
  const int n   = idx & (OUTF - 1);
  float acc = (float)bias[n];
  for (int s = 0; s < nsplit; ++s)
    acc += partial[(size_t)s * TOKENS * OUTF + idx];
  out[idx] = (_Float16)acc;
}

extern "C" void kernel_launch(void* const* d_in, const int* /*in_sizes*/, int /*n_in*/,
                              void* d_out, int /*out_size*/, void* d_ws, size_t ws_size,
                              hipStream_t stream) {
  const _Float16* x  = (const _Float16*)d_in[0];
  const uint32_t* qw = (const uint32_t*)d_in[1];
  const uint32_t* qz = (const uint32_t*)d_in[2];
  const _Float16* sc = (const _Float16*)d_in[3];
  // d_in[4] (g_idx) is the trivial arange//GROUPSIZE mapping -> computed inline.
  const _Float16* bs = (const _Float16*)d_in[5];
  _Float16* out = (_Float16*)d_out;
  float*    ws  = (float*)d_ws;

  const size_t slab = (size_t)TOKENS * OUTF * sizeof(float);  // 1 MB per split
  if (ws_size >= 8 * slab) {
    gptq_wmma_gemm<INF / 8><<<dim3(OUTF / 128, 8), 256, 0, stream>>>(x, qw, qz, sc, ws);
    gptq_reduce_bias<<<(TOKENS * OUTF) / 256, 256, 0, stream>>>(ws, bs, out, 8);
  } else {
    gptq_wmma_gemm<INF / 4><<<dim3(OUTF / 128, 4), 256, 0, stream>>>(x, qw, qz, sc, ws);
    gptq_reduce_bias<<<(TOKENS * OUTF) / 256, 256, 0, stream>>>(ws, bs, out, 4);
  }
}
